// Sampler_54090818126277
// MI455X (gfx1250) — compile-verified, matrix operand order checked by reading, not run on hardware
//
#include <hip/hip_runtime.h>
#include <hip/hip_bf16.h>
#include <stdint.h>

// Sizes from the reference: B=4096, L=512, H=32.
#define B_N 4096
#define L_N 512
#define H_N 32
#define BL  (B_N * L_N)          // 2,097,152 elements of v / each output
#define NW  ((L_N * H_N) * L_N)  // 8,388,608 elements of W1 viewed as [16384, 512]

#define LDS_ROW 520              // 512 shorts + 8-short pad (stride 260 words == 4 mod 64 banks)
#define LDS_PLANE (64 * LDS_ROW) // one hi or lo plane: 64 W1 rows (two sampler steps)
#define LDS_GRP (16 * LDS_ROW)   // one 16-row n-tile group within a plane (8320 shorts)

typedef __attribute__((ext_vector_type(16))) __bf16 v16bf;
typedef __attribute__((ext_vector_type(8)))  float  v8f;

// ---------- fp32 -> bf16 (round to nearest even) split helpers ----------
__device__ __forceinline__ unsigned short bf16_rne(float f) {
    uint32_t u = __float_as_uint(f);
    u += 0x7FFFu + ((u >> 16) & 1u);
    return (unsigned short)(u >> 16);
}
__device__ __forceinline__ float bf16_to_f32(unsigned short s) {
    return __uint_as_float(((uint32_t)s) << 16);
}

// ---------------- prep kernels: build v and split operands ----------------
__global__ void __launch_bounds__(256) prep_v(const float* __restrict__ mu,
                                              const float* __restrict__ log_var,
                                              const float* __restrict__ eps0,
                                              unsigned short* __restrict__ v_hi,
                                              unsigned short* __restrict__ v_lo) {
    int idx = blockIdx.x * 256 + threadIdx.x;
    if (idx >= BL) return;
    int b = idx >> 9;        // / 512
    int j = idx & 511;
    float val;
    if (j == 0) {
        // input0 = eps0 * exp(0.5*log_var[:,0]) + mu[:,0]
        val = eps0[b] * expf(0.5f * log_var[b * L_N]) + mu[b * L_N];
    } else {
        val = mu[b * L_N + j - 1];
    }
    unsigned short hi = bf16_rne(val);
    v_hi[idx] = hi;
    v_lo[idx] = bf16_rne(val - bf16_to_f32(hi));
}

__global__ void __launch_bounds__(256) prep_w1(const float* __restrict__ W1,
                                               unsigned short* __restrict__ w1_hi,
                                               unsigned short* __restrict__ w1_lo) {
    int idx = blockIdx.x * 256 + threadIdx.x;
    if (idx >= NW) return;
    float val = W1[idx];
    unsigned short hi = bf16_rne(val);
    w1_hi[idx] = hi;
    w1_lo[idx] = bf16_rne(val - bf16_to_f32(hi));
}

// ---------------- WMMA fragment loaders ----------------
// 16-bit A/B fragment per CDNA5 ISA layout for 16x16x32:
//   lane l (0..15):  row = l, khalf=0 -> K = {0..7} U {16..23}
//   lane l (16..31): row = l-16, khalf=1 -> K = {8..15} U {24..31}
// Both 8-element K runs are contiguous in row-major memory -> two 16B loads.
union Frag {
    v16bf v;
    uint4 q[2];
};

__device__ __forceinline__ void load_frag(Frag& f, const unsigned short* __restrict__ base,
                                          int row, int kb, int khalf) {
    const uint4* p = reinterpret_cast<const uint4*>(base + row * L_N + kb + khalf * 8);
    f.q[0] = p[0];   // K = kb + khalf*8 + (0..7)
    f.q[1] = p[2];   // K = kb + 16 + khalf*8 + (0..7)
}

// LDS fragment load: `base` is the per-lane plane pointer (already offset by
// m*LDS_ROW + khalf*8 + kb); GRP selects the 16-row n-tile group, so all DS
// offsets are compile-time constants < 64 KB -> constant-offset ds_load_b128.
template <int GRP>
__device__ __forceinline__ void load_frag_lds(Frag& f, const unsigned short* base) {
    f.q[0] = *reinterpret_cast<const uint4*>(base + GRP * LDS_GRP);
    f.q[1] = *reinterpret_cast<const uint4*>(base + GRP * LDS_GRP + 16);
}

// ---------------- main fused kernel ----------------
// Block: 256 threads = 8 wave32. Block handles 128 b-rows x TWO sampler steps
// (i0, i0+1). W1 hi/lo slices for both steps are staged once into LDS (130 KB)
// and read by all waves via ds_load_b128 (kills the 8x redundant L2 reads);
// handling two steps per block halves the per-FLOP A traffic. Each wave owns a
// 16(b) x 32(h) tile per step: 4 f32 accumulators, 12 bf16 WMMA per 32-wide
// K-chunk (hi*hi + hi*lo + lo*hi split for fp32-level accuracy).
// The same LDS buffer is reused (after a barrier) to stage the 128x64 H tile
// for the fused [2,H] contraction + bias + reparameterized sampling.
__global__ void __launch_bounds__(256) sampler_main(
    const unsigned short* __restrict__ v_hi, const unsigned short* __restrict__ v_lo,
    const unsigned short* __restrict__ w1_hi, const unsigned short* __restrict__ w1_lo,
    const float* __restrict__ b1, const float* __restrict__ W2,
    const float* __restrict__ b2, const float* __restrict__ eps,
    float* __restrict__ out) {

    extern __shared__ unsigned short smem[];   // 2 * LDS_PLANE shorts = 133,120 B

    const int lane    = threadIdx.x & 31;
    const int wave    = threadIdx.x >> 5;
    const int i0      = blockIdx.y * 2;                 // sampler steps i0, i0+1
    const int rowBase = blockIdx.x * 128 + wave * 16;
    const int m       = lane & 15;
    const int khalf   = lane >> 4;

    // ---- stage W1 hi/lo rows (i0*32 .. i0*32+63) into LDS ----
    {
        for (int q = threadIdx.x; q < 8192; q += 256) {
            int c      = q & 63;        // 16B chunk within a 512-short row
            int copyId = q >> 6;        // 0..127 row-copies
            int plane  = copyId >> 6;   // 0 = hi, 1 = lo
            int r      = copyId & 63;   // LDS row 0..63 -> global row i0*32 + r
            const unsigned short* src = (plane ? w1_lo : w1_hi) + (i0 * 32 + r) * L_N + c * 8;
            uint4 d = *reinterpret_cast<const uint4*>(src);
            *reinterpret_cast<uint4*>(smem + plane * LDS_PLANE + r * LDS_ROW + c * 8) = d;
        }
    }
    __syncthreads();

    // Per-lane plane base pointers. The lo-plane displacement is passed through
    // an empty asm so it stays an opaque, loop-invariant VGPR: otherwise LLVM
    // re-associates loBase into hiBase + 66,560B, which overflows the 16-bit DS
    // offset field and forces per-chunk v_add address materialization.
    const unsigned short* hiBase = smem + m * LDS_ROW + khalf * 8;
    unsigned loPlaneOff = LDS_PLANE;
    asm("" : "+v"(loPlaneOff));
    const unsigned short* loBase = hiBase + loPlaneOff;

    v8f a00 = {0.f,0.f,0.f,0.f,0.f,0.f,0.f,0.f};  // step i0, h cols 0..15
    v8f a01 = {0.f,0.f,0.f,0.f,0.f,0.f,0.f,0.f};  // step i0, h cols 16..31
    v8f a10 = {0.f,0.f,0.f,0.f,0.f,0.f,0.f,0.f};  // step i0+1, h cols 0..15
    v8f a11 = {0.f,0.f,0.f,0.f,0.f,0.f,0.f,0.f};  // step i0+1, h cols 16..31

    for (int kb = 0; kb < L_N; kb += 32) {
        Frag aHi, aLo;
        load_frag(aHi, v_hi, rowBase + m, kb, khalf);
        load_frag(aLo, v_lo, rowBase + m, kb, khalf);

        const unsigned short* pH = hiBase + kb;
        const unsigned short* pL = loBase + kb;
        Frag b00h, b00l, b01h, b01l, b10h, b10l, b11h, b11l;
        load_frag_lds<0>(b00h, pH);
        load_frag_lds<0>(b00l, pL);
        load_frag_lds<1>(b01h, pH);
        load_frag_lds<1>(b01l, pL);
        load_frag_lds<2>(b10h, pH);
        load_frag_lds<2>(b10l, pL);
        load_frag_lds<3>(b11h, pH);
        load_frag_lds<3>(b11l, pL);

        // fp32-accurate product via bf16 split: hi*hi + hi*lo + lo*hi
        a00 = __builtin_amdgcn_wmma_f32_16x16x32_bf16(false, aHi.v, false, b00h.v, (short)0, a00, false, false);
        a00 = __builtin_amdgcn_wmma_f32_16x16x32_bf16(false, aHi.v, false, b00l.v, (short)0, a00, false, false);
        a00 = __builtin_amdgcn_wmma_f32_16x16x32_bf16(false, aLo.v, false, b00h.v, (short)0, a00, false, false);
        a01 = __builtin_amdgcn_wmma_f32_16x16x32_bf16(false, aHi.v, false, b01h.v, (short)0, a01, false, false);
        a01 = __builtin_amdgcn_wmma_f32_16x16x32_bf16(false, aHi.v, false, b01l.v, (short)0, a01, false, false);
        a01 = __builtin_amdgcn_wmma_f32_16x16x32_bf16(false, aLo.v, false, b01h.v, (short)0, a01, false, false);
        a10 = __builtin_amdgcn_wmma_f32_16x16x32_bf16(false, aHi.v, false, b10h.v, (short)0, a10, false, false);
        a10 = __builtin_amdgcn_wmma_f32_16x16x32_bf16(false, aHi.v, false, b10l.v, (short)0, a10, false, false);
        a10 = __builtin_amdgcn_wmma_f32_16x16x32_bf16(false, aLo.v, false, b10h.v, (short)0, a10, false, false);
        a11 = __builtin_amdgcn_wmma_f32_16x16x32_bf16(false, aHi.v, false, b11h.v, (short)0, a11, false, false);
        a11 = __builtin_amdgcn_wmma_f32_16x16x32_bf16(false, aHi.v, false, b11l.v, (short)0, a11, false, false);
        a11 = __builtin_amdgcn_wmma_f32_16x16x32_bf16(false, aLo.v, false, b11h.v, (short)0, a11, false, false);
    }

    __syncthreads();   // everyone done reading W1 from LDS; safe to reuse as h buffer

    // Epilogue part 1: bias + leaky_relu(0.01), spill H tiles to LDS.
    // C/D layout: VGPR r, lanes 0-15 -> (M=r, N=lane); lanes 16-31 -> (M=r+8, N=lane-16)
    float* hbuf = reinterpret_cast<float*>(smem);   // [128 rows][65 floats] (pad -> conflict-free)
    const float b00 = b1[i0 * H_N + m];
    const float b01 = b1[i0 * H_N + 16 + m];
    const float b10 = b1[(i0 + 1) * H_N + m];
    const float b11 = b1[(i0 + 1) * H_N + 16 + m];
#pragma unroll
    for (int r = 0; r < 8; ++r) {
        int lr = wave * 16 + r + 8 * khalf;
        float h;
        h = a00[r] + b00; hbuf[lr * 65 + m]      = h > 0.f ? h : 0.01f * h;
        h = a01[r] + b01; hbuf[lr * 65 + 16 + m] = h > 0.f ? h : 0.01f * h;
        h = a10[r] + b10; hbuf[lr * 65 + 32 + m] = h > 0.f ? h : 0.01f * h;
        h = a11[r] + b11; hbuf[lr * 65 + 48 + m] = h > 0.f ? h : 0.01f * h;
    }
    __syncthreads();

    // Epilogue part 2: one (b, i) pair per thread.
    const int row  = threadIdx.x & 127;
    const int isel = threadIdx.x >> 7;      // 0 -> i0, 1 -> i0+1
    const int i    = i0 + isel;
    const int b    = blockIdx.x * 128 + row;
    float s0 = b2[i * 2 + 0];
    float s1 = b2[i * 2 + 1];
    const float* __restrict__ w2a = W2 + i * 2 * H_N;        // W2[i,0,:]
    const float* __restrict__ w2b = W2 + i * 2 * H_N + H_N;  // W2[i,1,:]
    const float* hp = hbuf + row * 65 + isel * 32;
#pragma unroll
    for (int hh = 0; hh < H_N; ++hh) {
        float hv = hp[hh];
        s0 += hv * w2a[hh];
        s1 += hv * w2b[hh];
    }
    float smp = eps[b * L_N + i] * expf(0.5f * s1) + s0;
    out[0 * BL + b * L_N + i] = s0;   // mus_resampled
    out[1 * BL + b * L_N + i] = s1;   // log_vars_resampled
    out[2 * BL + b * L_N + i] = smp;  // samples
}

// ---------------- launcher ----------------
extern "C" void kernel_launch(void* const* d_in, const int* in_sizes, int n_in,
                              void* d_out, int out_size, void* d_ws, size_t ws_size,
                              hipStream_t stream) {
    (void)in_sizes; (void)n_in; (void)out_size; (void)ws_size;
    const float* mu      = (const float*)d_in[0];
    const float* log_var = (const float*)d_in[1];
    const float* eps0    = (const float*)d_in[2];
    const float* eps     = (const float*)d_in[3];
    const float* W1      = (const float*)d_in[4];
    const float* b1      = (const float*)d_in[5];
    const float* W2      = (const float*)d_in[6];
    const float* b2      = (const float*)d_in[7];
    float* out = (float*)d_out;

    // Workspace layout (bf16 hi/lo splits), ~42 MB total, 16B-aligned slices.
    unsigned short* v_hi  = (unsigned short*)d_ws;
    unsigned short* v_lo  = v_hi + BL;
    unsigned short* w1_hi = v_lo + BL;
    unsigned short* w1_lo = w1_hi + NW;

    prep_v <<<BL / 256, 256, 0, stream>>>(mu, log_var, eps0, v_hi, v_lo);
    prep_w1<<<NW / 256, 256, 0, stream>>>(W1, w1_hi, w1_lo);

    dim3 grid(B_N / 128, L_N / 2);                 // 32 x 256 blocks, 8 wave32 each
    size_t smemBytes = 2u * LDS_PLANE * sizeof(unsigned short);  // 133,120 B
    sampler_main<<<grid, 256, smemBytes, stream>>>(v_hi, v_lo, w1_hi, w1_lo,
                                                   b1, W2, b2, eps, out);
}
